// GAT_23914377904690
// MI455X (gfx1250) — compile-verified, matrix operand order checked by reading, not run on hardware
//
#include <hip/hip_runtime.h>
#include <math.h>

#define N_NODES    50000
#define N_EDGES_IN 800000
#define EPRIME     (N_EDGES_IN + N_NODES)   // self loops appended
#define IN_DIM     128
#define HID        32
#define HEADS      4
#define EMB        64
#define NEG_SLOPE  0.2f
#define LN_EPS     1e-5f

typedef __attribute__((ext_vector_type(16))) _Float16 v16h;
typedef __attribute__((ext_vector_type(8)))  _Float16 v8h;
typedef __attribute__((ext_vector_type(4)))  _Float16 v4h;
typedef __attribute__((ext_vector_type(8)))  float    v8f;

// ---------------------------------------------------------------------------
// f32 -> f16 streaming conversion, 4 elements/thread (b128 in, b64 out)
// ---------------------------------------------------------------------------
__global__ void GAT_cvt_f16x4(const float* __restrict__ src,
                              _Float16* __restrict__ dst, int n4) {
  int i = blockIdx.x * blockDim.x + threadIdx.x;
  if (i >= n4) return;
  float4 v = ((const float4*)src)[i];
  v4h h = { (_Float16)v.x, (_Float16)v.y, (_Float16)v.z, (_Float16)v.w };
  ((v4h*)dst)[i] = h;
}

// ---------------------------------------------------------------------------
// Pack f32 weight B[K,N] (row-major) into WMMA B-operand lane layout, f16:
//   Bp[((tile*(K/32)+chunk)*32 + lane)*16 + j]
// where lane = group*16 + r holds column N = tile*16 + r and
//   K = chunk*32 + (j>>3)*16 + group*8 + (j&7).
// After packing, each lane's 16 B-halves per chunk are one 32B contiguous load.
// ---------------------------------------------------------------------------
template <int N, int K>
__global__ void GAT_pack_B(const float* __restrict__ B, _Float16* __restrict__ Bp) {
  constexpr int total = (N / 16) * (K / 32) * 512;
  int idx = blockIdx.x * blockDim.x + threadIdx.x;
  if (idx >= total) return;
  int j    = idx & 15;
  int lane = (idx >> 4) & 31;
  int rem  = idx >> 9;               // tile*(K/32) + chunk
  int chunk = rem % (K / 32);
  int tile  = rem / (K / 32);
  int group = lane >> 4, r = lane & 15;
  int kk = chunk * 32 + (j >> 3) * 16 + group * 8 + (j & 7);
  Bp[idx] = (_Float16)B[(size_t)kk * N + tile * 16 + r];
}

// ---------------------------------------------------------------------------
// WMMA GEMM: C[M,N](f32) = A[M,K](f16, row-major) @ B(pre-packed f16).
// One wave per 16x16 output tile; K-loop fully unrolled (K/32 WMMAs).
// ---------------------------------------------------------------------------
template <int N, int K>
__global__ void GAT_gemm_wmma(const _Float16* __restrict__ A,
                              const _Float16* __restrict__ Bp,
                              float* __restrict__ C, int M) {
  int wave   = (blockIdx.x * blockDim.x + threadIdx.x) >> 5;
  int lane   = threadIdx.x & 31;
  constexpr int tilesN = N >> 4;
  int totTiles = (M >> 4) * tilesN;
  if (wave >= totTiles) return;
  int tm = wave / tilesN;
  int tn = wave - tm * tilesN;
  int group = lane >> 4;     // 0: K-offset 0, 1: K-offset 8 (within 16-chunk)
  int r     = lane & 15;

  v8f acc = {};
  const _Float16* arow = A + (size_t)(tm * 16 + r) * K;  // lane holds row M=r
  const _Float16* brow = Bp + (((size_t)tn * (K / 32) * 32) + lane) * 16;
#pragma unroll
  for (int k0 = 0; k0 < K; k0 += 32) {
    // A f16 16x32 layout: elems 0..7  -> K = k0 + group*8 + t
    //                     elems 8..15 -> K = k0 + 16 + group*8 + t
    v8h a0 = *(const v8h*)(arow + k0 + group * 8);
    v8h a1 = *(const v8h*)(arow + k0 + group * 8 + 16);
    v16h a;
#pragma unroll
    for (int t = 0; t < 8; ++t) { a[t] = a0[t]; a[t + 8] = a1[t]; }
    // B: pre-packed, contiguous 16 halves per lane per chunk
    v16h b = *(const v16h*)(brow + (size_t)(k0 / 32) * 512);
    acc = __builtin_amdgcn_wmma_f32_16x16x32_f16(false, a, false, b,
                                                 (short)0, acc, false, false);
  }
  // D layout: VGPR v -> row M = v + 8*group, col N = r
  float* crow = C + (size_t)(tm * 16 + 8 * group) * N + tn * 16 + r;
#pragma unroll
  for (int v = 0; v < 8; ++v) crow[(size_t)v * N] = acc[v];
}

// ---------------------------------------------------------------------------
// Per-(node, head) attention coefficients: a_s = h . att_src, a_d = h . att_dst
// ---------------------------------------------------------------------------
__global__ void GAT_att_coef(const float* __restrict__ h,
                             const float* __restrict__ att_s,
                             const float* __restrict__ att_d,
                             float* __restrict__ as_, float* __restrict__ ad_,
                             int n, int H, int Cc) {
  int t = blockIdx.x * blockDim.x + threadIdx.x;
  if (t >= n * H) return;
  int node = t / H, hh = t - node * H;
  const float* hp = h + (size_t)node * H * Cc + (size_t)hh * Cc;
  const float* sp = att_s + hh * Cc;
  const float* dp = att_d + hh * Cc;
  float accs = 0.f, accd = 0.f;
  for (int c = 0; c < Cc; ++c) { float v = hp[c]; accs += v * sp[c]; accd += v * dp[c]; }
  as_[t] = accs;
  ad_[t] = accd;
}

__global__ void GAT_fill_u32(unsigned* __restrict__ p, unsigned v, long long n) {
  long long i = (long long)blockIdx.x * blockDim.x + threadIdx.x;
  if (i < n) p[i] = v;
}

// monotonic uint key for float atomic-max
__device__ __forceinline__ unsigned GAT_fkey(float x) {
  unsigned u = __float_as_uint(x);
  return (u & 0x80000000u) ? ~u : (u | 0x80000000u);
}
__device__ __forceinline__ float GAT_unfkey(unsigned k) {
  return (k & 0x80000000u) ? __uint_as_float(k & 0x7fffffffu) : __uint_as_float(~k);
}

__device__ __forceinline__ void GAT_edge_sd(const int* __restrict__ ei, int e,
                                            int& s, int& d) {
  if (e < N_EDGES_IN) { s = ei[e]; d = ei[N_EDGES_IN + e]; }
  else                { s = e - N_EDGES_IN; d = s; }          // self loop
}

// pass 1: logits + segment max (uint-keyed atomicMax)
__global__ void GAT_edge_logits_max(const int* __restrict__ ei,
                                    const float* __restrict__ as_,
                                    const float* __restrict__ ad_,
                                    float* __restrict__ eb,
                                    unsigned* __restrict__ mk, int H) {
  int t = blockIdx.x * blockDim.x + threadIdx.x;
  if (t >= EPRIME * H) return;
  int e = t / H, hh = t - e * H;
  int s, d; GAT_edge_sd(ei, e, s, d);
  float x = as_[s * H + hh] + ad_[d * H + hh];
  x = (x > 0.f) ? x : NEG_SLOPE * x;                 // leaky relu
  eb[t] = x;
  atomicMax(&mk[d * H + hh], GAT_fkey(x));
}

// pass 2: exp(e - m[dst]) + segment sum
__global__ void GAT_edge_exp_sum(const int* __restrict__ ei,
                                 float* __restrict__ eb,
                                 const unsigned* __restrict__ mk,
                                 float* __restrict__ den, int H) {
  int t = blockIdx.x * blockDim.x + threadIdx.x;
  if (t >= EPRIME * H) return;
  int e = t / H, hh = t - e * H;
  int s, d; GAT_edge_sd(ei, e, s, d);
  float ex = __expf(eb[t] - GAT_unfkey(mk[d * H + hh]));
  eb[t] = ex;
  atomicAdd(&den[d * H + hh], ex);
}

// pass 3: agg[dst] += h[src] * alpha. One block per edge, one thread per channel.
__global__ void GAT_edge_scatter(const int* __restrict__ ei,
                                 const float* __restrict__ hsrc,
                                 const float* __restrict__ eb,
                                 const float* __restrict__ den,
                                 float* __restrict__ agg, int H, int Cc) {
  int e = blockIdx.x;
  int t = threadIdx.x;              // t in [0, H*Cc)
  int hh = t / Cc;
  int s, d; GAT_edge_sd(ei, e, s, d);
  float alpha = eb[(size_t)e * H + hh] / den[d * H + hh];
  atomicAdd(&agg[(size_t)d * H * Cc + t], hsrc[(size_t)s * H * Cc + t] * alpha);
}

// bias + layernorm(128) + relu, one wave per node (4 elems/lane, wave32 shfl).
// Output written as f16 (sole consumer is the layer-2 WMMA GEMM A-operand).
__global__ void GAT_bias_ln_relu128(const float* __restrict__ agg,
                                    const float* __restrict__ b,
                                    const float* __restrict__ g,
                                    const float* __restrict__ be,
                                    _Float16* __restrict__ out) {
  int wave = (blockIdx.x * blockDim.x + threadIdx.x) >> 5;
  int lane = threadIdx.x & 31;
  if (wave >= N_NODES) return;
  const float* p = agg + (size_t)wave * 128;
  float v[4]; float sum = 0.f;
#pragma unroll
  for (int j = 0; j < 4; ++j) { int i = lane + 32 * j; v[j] = p[i] + b[i]; sum += v[j]; }
#pragma unroll
  for (int o = 16; o > 0; o >>= 1) sum += __shfl_xor(sum, o, 32);
  float mu = sum * (1.f / 128.f);
  float sq = 0.f;
#pragma unroll
  for (int j = 0; j < 4; ++j) { float dl = v[j] - mu; sq += dl * dl; }
#pragma unroll
  for (int o = 16; o > 0; o >>= 1) sq += __shfl_xor(sq, o, 32);
  float rs = rsqrtf(sq * (1.f / 128.f) + LN_EPS);
  _Float16* q = out + (size_t)wave * 128;
#pragma unroll
  for (int j = 0; j < 4; ++j) {
    int i = lane + 32 * j;
    q[i] = (_Float16)fmaxf((v[j] - mu) * rs * g[i] + be[i], 0.f);
  }
}

// bias + layernorm(64), no relu, final output. One wave per node (2 elems/lane).
__global__ void GAT_bias_ln64(const float* __restrict__ agg,
                              const float* __restrict__ b,
                              const float* __restrict__ g,
                              const float* __restrict__ be,
                              float* __restrict__ out) {
  int wave = (blockIdx.x * blockDim.x + threadIdx.x) >> 5;
  int lane = threadIdx.x & 31;
  if (wave >= N_NODES) return;
  const float* p = agg + (size_t)wave * 64;
  float v0 = p[lane] + b[lane];
  float v1 = p[lane + 32] + b[lane + 32];
  float sum = v0 + v1;
#pragma unroll
  for (int o = 16; o > 0; o >>= 1) sum += __shfl_xor(sum, o, 32);
  float mu = sum * (1.f / 64.f);
  float d0 = v0 - mu, d1 = v1 - mu;
  float sq = d0 * d0 + d1 * d1;
#pragma unroll
  for (int o = 16; o > 0; o >>= 1) sq += __shfl_xor(sq, o, 32);
  float rs = rsqrtf(sq * (1.f / 64.f) + LN_EPS);
  float* q = out + (size_t)wave * 64;
  q[lane]      = d0 * rs * g[lane]      + be[lane];
  q[lane + 32] = d1 * rs * g[lane + 32] + be[lane + 32];
}

// ---------------------------------------------------------------------------
extern "C" void kernel_launch(void* const* d_in, const int* in_sizes, int n_in,
                              void* d_out, int out_size, void* d_ws, size_t ws_size,
                              hipStream_t stream) {
  const float* x        = (const float*)d_in[0];
  const int*   ei       = (const int*)  d_in[1];   // [2, 800000]
  const float* W1       = (const float*)d_in[2];
  const float* att_src1 = (const float*)d_in[3];
  const float* att_dst1 = (const float*)d_in[4];
  const float* b1       = (const float*)d_in[5];
  const float* gamma1   = (const float*)d_in[6];
  const float* beta1    = (const float*)d_in[7];
  const float* W2       = (const float*)d_in[8];
  const float* att_src2 = (const float*)d_in[9];
  const float* att_dst2 = (const float*)d_in[10];
  const float* b2       = (const float*)d_in[11];
  const float* gamma2   = (const float*)d_in[12];
  const float* beta2    = (const float*)d_in[13];
  float* out = (float*)d_out;

  // workspace carving
  size_t off = 0;
  auto alloc = [&](size_t bytes) -> char* {
    char* p = (char*)d_ws + off;
    off += (bytes + 255) & ~(size_t)255;
    return p;
  };
  _Float16* xh   = (_Float16*)alloc(sizeof(_Float16) * (size_t)N_NODES * 128);
  _Float16* B1p  = (_Float16*)alloc(sizeof(_Float16) * (128 / 16) * (128 / 32) * 512);
  _Float16* B2p  = (_Float16*)alloc(sizeof(_Float16) * (64 / 16) * (128 / 32) * 512);
  float*    h1   = (float*)   alloc(sizeof(float) * (size_t)N_NODES * 128);
  float*    as1  = (float*)   alloc(sizeof(float) * (size_t)N_NODES * HEADS);
  float*    ad1  = (float*)   alloc(sizeof(float) * (size_t)N_NODES * HEADS);
  unsigned* mk1  = (unsigned*)alloc(sizeof(unsigned) * (size_t)N_NODES * HEADS);
  float*    den1 = (float*)   alloc(sizeof(float) * (size_t)N_NODES * HEADS);
  float*    eb1  = (float*)   alloc(sizeof(float) * (size_t)EPRIME * HEADS);
  float*    agg1 = (float*)   alloc(sizeof(float) * (size_t)N_NODES * 128);
  _Float16* h1ph = (_Float16*)alloc(sizeof(_Float16) * (size_t)N_NODES * 128);
  float*    h2   = (float*)   alloc(sizeof(float) * (size_t)N_NODES * 64);
  float*    as2  = (float*)   alloc(sizeof(float) * (size_t)N_NODES);
  float*    ad2  = (float*)   alloc(sizeof(float) * (size_t)N_NODES);
  unsigned* mk2  = (unsigned*)alloc(sizeof(unsigned) * (size_t)N_NODES);
  float*    den2 = (float*)   alloc(sizeof(float) * (size_t)N_NODES);
  float*    eb2  = (float*)   alloc(sizeof(float) * (size_t)EPRIME);
  float*    agg2 = (float*)   alloc(sizeof(float) * (size_t)N_NODES * 64);

  const unsigned NEG_INF_KEY = 0x007FFFFFu;  // fkey(-inf)
  auto blocks = [](long long n, int bs) { return (unsigned)((n + bs - 1) / bs); };

  // ---- operand pre-conversion / pre-packing to f16 ----
  GAT_cvt_f16x4<<<blocks((long long)N_NODES * 128 / 4, 256), 256, 0, stream>>>(x, xh, N_NODES * 128 / 4);
  GAT_pack_B<128, 128><<<blocks((128 / 16) * (128 / 32) * 512, 256), 256, 0, stream>>>(W1, B1p);
  GAT_pack_B<64, 128><<<blocks((64 / 16) * (128 / 32) * 512, 256), 256, 0, stream>>>(W2, B2p);

  // ---- layer 1 ----
  {
    long long waves = ((long long)N_NODES / 16) * (128 / 16);   // 25000
    GAT_gemm_wmma<128, 128><<<blocks(waves * 32, 256), 256, 0, stream>>>(xh, B1p, h1, N_NODES);
  }
  GAT_att_coef<<<blocks((long long)N_NODES * HEADS, 256), 256, 0, stream>>>(
      h1, att_src1, att_dst1, as1, ad1, N_NODES, HEADS, HID);
  GAT_fill_u32<<<blocks((long long)N_NODES * HEADS, 256), 256, 0, stream>>>(mk1, NEG_INF_KEY, (long long)N_NODES * HEADS);
  GAT_fill_u32<<<blocks((long long)N_NODES * HEADS, 256), 256, 0, stream>>>((unsigned*)den1, 0u, (long long)N_NODES * HEADS);
  GAT_fill_u32<<<blocks((long long)N_NODES * 128, 256), 256, 0, stream>>>((unsigned*)agg1, 0u, (long long)N_NODES * 128);
  GAT_edge_logits_max<<<blocks((long long)EPRIME * HEADS, 256), 256, 0, stream>>>(ei, as1, ad1, eb1, mk1, HEADS);
  GAT_edge_exp_sum<<<blocks((long long)EPRIME * HEADS, 256), 256, 0, stream>>>(ei, eb1, mk1, den1, HEADS);
  GAT_edge_scatter<<<EPRIME, HEADS * HID, 0, stream>>>(ei, h1, eb1, den1, agg1, HEADS, HID);
  GAT_bias_ln_relu128<<<blocks((long long)N_NODES * 32, 256), 256, 0, stream>>>(agg1, b1, gamma1, beta1, h1ph);

  // ---- layer 2 ----
  {
    long long waves = ((long long)N_NODES / 16) * (64 / 16);    // 12500
    GAT_gemm_wmma<64, 128><<<blocks(waves * 32, 256), 256, 0, stream>>>(h1ph, B2p, h2, N_NODES);
  }
  GAT_att_coef<<<blocks((long long)N_NODES, 256), 256, 0, stream>>>(
      h2, att_src2, att_dst2, as2, ad2, N_NODES, 1, EMB);
  GAT_fill_u32<<<blocks((long long)N_NODES, 256), 256, 0, stream>>>(mk2, NEG_INF_KEY, (long long)N_NODES);
  GAT_fill_u32<<<blocks((long long)N_NODES, 256), 256, 0, stream>>>((unsigned*)den2, 0u, (long long)N_NODES);
  GAT_fill_u32<<<blocks((long long)N_NODES * 64, 256), 256, 0, stream>>>((unsigned*)agg2, 0u, (long long)N_NODES * 64);
  GAT_edge_logits_max<<<blocks((long long)EPRIME, 256), 256, 0, stream>>>(ei, as2, ad2, eb2, mk2, 1);
  GAT_edge_exp_sum<<<blocks((long long)EPRIME, 256), 256, 0, stream>>>(ei, eb2, mk2, den2, 1);
  GAT_edge_scatter<<<EPRIME, EMB, 0, stream>>>(ei, h2, eb2, den2, agg2, 1, EMB);
  GAT_bias_ln64<<<blocks((long long)N_NODES * 32, 256), 256, 0, stream>>>(agg2, b2, gamma2, beta2, out);
}
